// LSTMAE_17300128268671
// MI455X (gfx1250) — compile-verified, matrix operand order checked by reading, not run on hardware
//
#include <hip/hip_runtime.h>
#include <hip/hip_bf16.h>

typedef __attribute__((ext_vector_type(8)))  __bf16 v8bf;
typedef __attribute__((ext_vector_type(16))) __bf16 v16bf;
typedef __attribute__((ext_vector_type(8)))  float  v8f;

#define B_TOT 512
#define T_TOT 512
#define STRIDE 64   // all activation buffers are [B, T, 64] bf16, zero-padded in K

#define LOG2E   1.4426950408889634f
#define LOG2E2  2.8853900817779268f

// Branchless transcendentals on the serial recurrence path.
__device__ __forceinline__ float fast_tanh(float x) {
#if __has_builtin(__builtin_amdgcn_tanhf)
    return __builtin_amdgcn_tanhf(x);          // V_TANH_F32 (CDNA5 TRANS op)
#else
    // tanh(x) = 1 - 2/(exp2(2*log2e*x) + 1); saturates correctly at +-inf
    float e = __builtin_amdgcn_exp2f(x * LOG2E2);
    return 1.0f - 2.0f * __builtin_amdgcn_rcpf(e + 1.0f);
#endif
}

__device__ __forceinline__ float fast_sig(float x) {
    // sigmoid(x) = 1/(1 + exp2(-x*log2e)) : one v_exp + one v_rcp, no branches
    return __builtin_amdgcn_rcpf(1.0f + __builtin_amdgcn_exp2f(-x * LOG2E));
}

__device__ __forceinline__ v16bf join8(v8bf lo, v8bf hi) {
    return __builtin_shufflevector(lo, hi, 0,1,2,3,4,5,6,7,8,9,10,11,12,13,14,15);
}

// One workgroup = one 16-row batch tile; wave w owns hidden columns [16w,16w+16).
// Weights are held in VGPRs (bf16 WMMA B-fragments) for the entire kernel.
// One LDS double-buffer + one barrier per timestep for the h D->A transpose.
template<int DIN, int H, bool STORE_SEQ, bool WRITE_HLAST, bool F32OUT>
__global__ __launch_bounds__(32 * (H / 16)) void lstm_layer(
    const __bf16* __restrict__ xin,          // [B,T,64] bf16 (K-padded with zeros)
    const float*  __restrict__ Wih,          // [4H, DIN] f32
    const float*  __restrict__ Whh,          // [4H, H]   f32
    const float*  __restrict__ bih,          // [4H]
    const float*  __restrict__ bhh,          // [4H]
    __bf16*       __restrict__ outseq,       // [B,T,64] bf16 (or null)
    __bf16*       __restrict__ hlast,        // [B,H] bf16 (or null)
    float*        __restrict__ f32out)       // [B,T,64] f32 (or null)
{
    constexpr int H16  = H / 16;            // waves per workgroup
    constexpr int NT   = 32 * H16;
    constexpr int KX   = (DIN + 31) & ~31;  // K padded to 32 for WMMA
    constexpr int KH   = (H + 31) & ~31;
    constexpr int KX32 = KX / 32;
    constexpr int KH32 = KH / 32;
    constexpr int HPAD = KH;

    __shared__ __bf16 s_h[2][16 * HPAD];    // double-buffered h (A-layout source)

    const int tid  = threadIdx.x;
    const int w    = tid >> 5;              // wave id = hidden-column block
    const int lane = tid & 31;
    const int n    = lane & 15;             // N col within tile / A-matrix row M
    const int hi   = lane >> 4;             // half-wave selector (K chunk select)
    const int b0   = blockIdx.x * 16;

    // zero both h buffers (covers h0 = 0 and permanent zero K-padding)
    for (int i = tid; i < 2 * 16 * HPAD; i += NT)
        ((__bf16*)s_h)[i] = (__bf16)0.0f;

    // ---- preload weight fragments into registers (bf16, ISA B-layout) ----
    v16bf bwx[KX32][4];
    v16bf bwh[KH32][4];
    float bias[4];
    #pragma unroll
    for (int q = 0; q < 4; ++q) {           // gate order i,f,g,o
        const int gc = q * H + 16 * w + n;  // gate column = weight row
        bias[q] = bih[gc] + bhh[gc];
        #pragma unroll
        for (int s = 0; s < KX32; ++s) {
            v16bf v;
            #pragma unroll
            for (int j = 0; j < 8; ++j) {
                int k0 = s * 32 + 8 * hi + j;
                int k1 = s * 32 + 16 + 8 * hi + j;
                v[j]     = (k0 < DIN) ? (__bf16)Wih[gc * DIN + k0] : (__bf16)0.0f;
                v[8 + j] = (k1 < DIN) ? (__bf16)Wih[gc * DIN + k1] : (__bf16)0.0f;
            }
            bwx[s][q] = v;
        }
        #pragma unroll
        for (int s = 0; s < KH32; ++s) {
            v16bf v;
            #pragma unroll
            for (int j = 0; j < 8; ++j) {
                int k0 = s * 32 + 8 * hi + j;
                int k1 = s * 32 + 16 + 8 * hi + j;
                v[j]     = (k0 < H) ? (__bf16)Whh[gc * H + k0] : (__bf16)0.0f;
                v[8 + j] = (k1 < H) ? (__bf16)Whh[gc * H + k1] : (__bf16)0.0f;
            }
            bwh[s][q] = v;
        }
    }

    v8f c;
    #pragma unroll
    for (int r = 0; r < 8; ++r) c[r] = 0.0f;

    // per-lane x row base (A-matrix row M = n)
    const __bf16* xrow = xin + (size_t)(b0 + n) * T_TOT * STRIDE;

    // prefetch x A-fragments for t = 0 (two contiguous 16B chunks per K-step)
    v16bf ax[KX32];
    #pragma unroll
    for (int s = 0; s < KX32; ++s) {
        const __bf16* pc = xrow + s * 32 + 8 * hi;
        ax[s] = join8(*(const v8bf*)pc, *(const v8bf*)(pc + 16));
    }

    __syncthreads();

    int p = 0;
    for (int t = 0; t < T_TOT; ++t) {
        // issue ALL h-fragment LDS loads first so their latency hides under the
        // x-side WMMAs (distinct destination registers -> no ds/wmma chaining)
        v16bf ah[KH32];
        #pragma unroll
        for (int s = 0; s < KH32; ++s) {
            const __bf16* pc = &s_h[p][n * HPAD + s * 32 + 8 * hi];
            ah[s] = join8(*(const v8bf*)pc, *(const v8bf*)(pc + 16));
        }

        v8f acc[4];
        #pragma unroll
        for (int q = 0; q < 4; ++q) {
            #pragma unroll
            for (int r = 0; r < 8; ++r) acc[q][r] = bias[q];
        }
        // gates += x_t @ Wih^T
        #pragma unroll
        for (int s = 0; s < KX32; ++s) {
            #pragma unroll
            for (int q = 0; q < 4; ++q)
                acc[q] = __builtin_amdgcn_wmma_f32_16x16x32_bf16(
                    false, ax[s], false, bwx[s][q], (short)0, acc[q], false, false);
        }
        // branchless prefetch of next timestep's x while gates are in flight
        {
            const int tn = (t + 1 < T_TOT) ? (t + 1) : t;
            const __bf16* xr = xrow + (size_t)tn * STRIDE;
            #pragma unroll
            for (int s = 0; s < KX32; ++s) {
                const __bf16* pc = xr + s * 32 + 8 * hi;
                ax[s] = join8(*(const v8bf*)pc, *(const v8bf*)(pc + 16));
            }
        }
        // gates += h_{t-1} @ Whh^T
        #pragma unroll
        for (int s = 0; s < KH32; ++s) {
            #pragma unroll
            for (int q = 0; q < 4; ++q)
                acc[q] = __builtin_amdgcn_wmma_f32_16x16x32_bf16(
                    false, ah[s], false, bwh[s][q], (short)0, acc[q], false, false);
        }
        // elementwise cell update; D-layout element (r,lane) -> row r+8*hi, col 16w+n
        #pragma unroll
        for (int r = 0; r < 8; ++r) {
            float iv = fast_sig(acc[0][r]);
            float fv = fast_sig(acc[1][r]);
            float gv = fast_tanh(acc[2][r]);
            float ov = fast_sig(acc[3][r]);
            float cv = fv * c[r] + iv * gv;
            c[r] = cv;
            float hv = ov * fast_tanh(cv);
            s_h[p ^ 1][(r + 8 * hi) * HPAD + 16 * w + n] = (__bf16)hv;
            if constexpr (F32OUT)
                f32out[((size_t)(b0 + r + 8 * hi) * T_TOT + t) * STRIDE + 16 * w + n] = hv;
        }
        __syncthreads();   // single barrier per step (double-buffered LDS)
        if constexpr (STORE_SEQ) {
            constexpr int CH = 16 * HPAD / 8;
            for (int ch = tid; ch < CH; ch += NT) {
                int m  = ch / (HPAD / 8);
                int cc = ch % (HPAD / 8);
                v8bf vv = *(const v8bf*)&s_h[p ^ 1][m * HPAD + cc * 8];
                *(v8bf*)&outseq[((size_t)(b0 + m) * T_TOT + t) * STRIDE + cc * 8] = vv;
            }
        }
        p ^= 1;
    }

    if constexpr (WRITE_HLAST) {
        for (int i = tid; i < 16 * H; i += NT) {
            int m = i / H, col = i % H;
            hlast[(size_t)(b0 + m) * H + col] = s_h[p][m * HPAD + col];
        }
    }
}

__global__ void f32_to_bf16_kernel(const float* __restrict__ in,
                                   __bf16* __restrict__ out, int nelem) {
    int i = blockIdx.x * blockDim.x + threadIdx.x;
    if (i < nelem) out[i] = (__bf16)in[i];
}

// x_enc[b,t,k] = last_h[b,k] for k<16, 0 for 16<=k<32 (d1 reads K padded to 32)
__global__ void broadcast_latent_kernel(const __bf16* __restrict__ hlast,
                                        __bf16* __restrict__ out) {
    int i = blockIdx.x * blockDim.x + threadIdx.x;  // over B*T*32
    int b = i / (T_TOT * 32);
    int r = i % (T_TOT * 32);
    int t = r / 32;
    int k = r % 32;
    out[((size_t)b * T_TOT + t) * STRIDE + k] =
        (k < 16) ? hlast[(size_t)b * 16 + k] : (__bf16)0.0f;
}

extern "C" void kernel_launch(void* const* d_in, const int* in_sizes, int n_in,
                              void* d_out, int out_size, void* d_ws, size_t ws_size,
                              hipStream_t stream) {
    (void)in_sizes; (void)n_in; (void)out_size; (void)ws_size;
    const float* x = (const float*)d_in[0];
    const float* W[6][4];
    for (int l = 0; l < 6; ++l)
        for (int j = 0; j < 4; ++j)
            W[l][j] = (const float*)d_in[1 + 4 * l + j];

    const size_t ACT = (size_t)B_TOT * T_TOT * STRIDE;   // elems per activation buf
    __bf16* actA  = (__bf16*)d_ws;            // ws usage: 2*32MB + 16KB
    __bf16* actB  = actA + ACT;
    __bf16* hlast = actB + ACT;
    float*  out   = (float*)d_out;

    f32_to_bf16_kernel<<<(int)(ACT / 256), 256, 0, stream>>>(x, actA, (int)ACT);

    // encoder
    lstm_layer<64, 48, true,  false, false><<<B_TOT / 16,  96, 0, stream>>>(
        actA, W[0][0], W[0][1], W[0][2], W[0][3], actB, nullptr, nullptr);
    lstm_layer<48, 32, true,  false, false><<<B_TOT / 16,  64, 0, stream>>>(
        actB, W[1][0], W[1][1], W[1][2], W[1][3], actA, nullptr, nullptr);
    lstm_layer<32, 16, false, true,  false><<<B_TOT / 16,  32, 0, stream>>>(
        actA, W[2][0], W[2][1], W[2][2], W[2][3], nullptr, hlast, nullptr);

    // repeat latent across time
    broadcast_latent_kernel<<<B_TOT * T_TOT * 32 / 256, 256, 0, stream>>>(hlast, actB);

    // decoder
    lstm_layer<16, 32, true,  false, false><<<B_TOT / 16,  64, 0, stream>>>(
        actB, W[3][0], W[3][1], W[3][2], W[3][3], actA, nullptr, nullptr);
    lstm_layer<32, 48, true,  false, false><<<B_TOT / 16,  96, 0, stream>>>(
        actA, W[4][0], W[4][1], W[4][2], W[4][3], actB, nullptr, nullptr);
    lstm_layer<48, 64, false, false, true ><<<B_TOT / 16, 128, 0, stream>>>(
        actB, W[5][0], W[5][1], W[5][2], W[5][3], nullptr, nullptr, out);
}
